// NonLocalMP_13898514170057
// MI455X (gfx1250) — matrix operand order repaired and checked by reading, not compiled, not run on hardware
//
#include <hip/hip_runtime.h>
#include <hip/hip_bf16.h>
#include <stdint.h>

typedef __attribute__((ext_vector_type(16))) __bf16        v16bf;
typedef __attribute__((ext_vector_type(8)))  float         v8f;
typedef __attribute__((ext_vector_type(8)))  unsigned int  v8u;

#define NEG_SLOPE 0.2f
#define BN_EPS    1e-5f

// Problem sizes (fixed by reference): B=256, V=16, D=512
#define NB  256
#define NV  16
#define ND  512
#define NROWS (NB*NV)          // 4096
#define LDP 520                // padded LDS row stride (bf16 elems): 260 dwords -> bank shift 4/row

__device__ __forceinline__ float lrelu(float x) { return x >= 0.f ? x : NEG_SLOPE * x; }

__device__ __forceinline__ unsigned short f2bf(float f) {
    unsigned int u = __builtin_bit_cast(unsigned int, f);
    u += 0x7fffu + ((u >> 16) & 1u);        // round-to-nearest-even
    return (unsigned short)(u >> 16);
}

// ---- WMMA fragment loaders (bf16, 16x16x32) -------------------------------
// A: 16x32, row-major source (stride ld elements). Contiguous k order (this code's choice, same as load_b_frag):
// lanes 0-15 hold K 0..15 (pairs per VGPR); lanes 16-31 hold K 16..31.
// Per-lane data is one contiguous 32B run.
__device__ __forceinline__ v16bf load_a_frag(const unsigned short* p, int ld, int k0) {
    int lane = threadIdx.x & 31;
    int m    = lane & 15;
    int half = lane >> 4;
    v8u a;
#pragma unroll
    for (int v = 0; v < 8; ++v) {
        int k = k0 + (v << 1) + (half << 4);
        a[v] = *(const unsigned int*)(p + m * ld + k);
    }
    return __builtin_bit_cast(v16bf, a);
}

// B: 32x16 taken from N-major (transposed) weights Wt[n][k], stride ldb.
// Lane n holds column n; lanes 0-15 cover K k0..k0+15, lanes 16-31 K k0+16..k0+31.
__device__ __forceinline__ v16bf load_b_frag(const unsigned short* wt, int ldb, int k0) {
    int lane = threadIdx.x & 31;
    int n    = lane & 15;
    int half = lane >> 4;
    return *(const v16bf*)(wt + n * ldb + k0 + (half << 4));
}

// ---- prep kernels ---------------------------------------------------------
__global__ void convert_kernel(const float* __restrict__ src, unsigned short* __restrict__ dst, int n) {
    int i = blockIdx.x * 256 + threadIdx.x;
    if (i < n) dst[i] = f2bf(src[i]);
}

// dst[n*K + k] = src[(k+rowoff)*N + n]   (K x N source, N-major bf16 dest)
__global__ void transpose_kernel(const float* __restrict__ src, unsigned short* __restrict__ dst,
                                 int K, int N, int rowoff) {
    int i = blockIdx.x * 256 + threadIdx.x;
    if (i < N * K) {
        int n = i / K, k = i % K;
        dst[i] = f2bf(src[(k + rowoff) * N + n]);
    }
}

// ---- generic bf16 GEMM: C(M x N, f32) = A(M x K) * Bt(N x K)^T ------------
// one wave per 16x64 tile; grid = (M/16, N/64), block = 32
__global__ void gemm16x64(const unsigned short* __restrict__ A, int lda,
                          const unsigned short* __restrict__ Bt, int ldb,
                          float* __restrict__ C, int ldc, int K) {
    int mt = blockIdx.x, nt = blockIdx.y;
    const unsigned short* Ap = A + (size_t)mt * 16 * lda;
    v8f acc[4] = {};
    for (int k0 = 0; k0 < K; k0 += 32) {
        v16bf a = load_a_frag(Ap, lda, k0);
#pragma unroll
        for (int s = 0; s < 4; ++s) {
            v16bf b = load_b_frag(Bt + (size_t)(nt * 64 + s * 16) * ldb, ldb, k0);
            acc[s] = __builtin_amdgcn_wmma_f32_16x16x32_bf16(false, a, false, b,
                                                             (short)0, acc[s], false, false);
        }
    }
    int lane = threadIdx.x & 31;
    int n    = lane & 15;
    int mo   = (lane >> 4) * 8;
    float* Cp = C + (size_t)(mt * 16 + mo) * ldc + nt * 64 + n;
#pragma unroll
    for (int s = 0; s < 4; ++s)
#pragma unroll
        for (int r = 0; r < 8; ++r)
            Cp[(size_t)r * ldc + s * 16] = acc[s][r];
}

// ---- fused relation MLP (layers 2,3) + neighbor sum -----------------------
// 1 workgroup per batch item b; 8 waves (256 thr). Two i-tiles per pass so
// each W2/W3 B-fragment feeds two WMMAs (halves L2 weight traffic).
// Per tile (i=t+u): h1[j][:] = lrelu(P[b,i]+Q[b,j]+b1) in LDS (bf16, padded),
// layer-2 GEMM -> LDS, layer-3 GEMM -> lrelu -> sum over the 16 rows (=j)
// in accumulator fragments -> S[b,i].
__global__ __launch_bounds__(256) void relation_kernel(
    const float* __restrict__ P, const float* __restrict__ Q,
    const unsigned short* __restrict__ Wt2, const unsigned short* __restrict__ Wt3,
    const float* __restrict__ b1, const float* __restrict__ b2, const float* __restrict__ b3,
    unsigned short* __restrict__ S) {
    __shared__ __align__(16) unsigned short h1s[2][16 * LDP];   // ~32.5 KB
    __shared__ __align__(16) unsigned short h2s[2][16 * LDP];   // ~32.5 KB
    int b    = blockIdx.x;
    int tid  = threadIdx.x;
    int w    = tid >> 5;
    int lane = tid & 31;
    int n    = lane & 15;
    int mo   = (lane >> 4) * 8;

    for (int t = 0; t < NV; t += 2) {
        // ---- h1 for both tiles: rows j=0..15, cols 0..511
        for (int idx = tid; idx < 2 * 16 * ND; idx += 256) {
            int u = idx >> 13;            // tile index (16*512 = 8192 per tile)
            int j = (idx >> 9) & 15;
            int c = idx & (ND - 1);
            float v = P[(size_t)(b * NV + t + u) * ND + c] +
                      Q[(size_t)(b * NV + j) * ND + c] + b1[c];
            h1s[u][j * LDP + c] = f2bf(lrelu(v));
        }
        __syncthreads();

        // ---- layer 2: h2 = lrelu(h1 @ W2 + b2); wave w owns cols [w*64, w*64+64)
        {
            v8f acc[2][4] = {};
            for (int k0 = 0; k0 < ND; k0 += 32) {
                v16bf a0 = load_a_frag(h1s[0], LDP, k0);
                v16bf a1 = load_a_frag(h1s[1], LDP, k0);
#pragma unroll
                for (int s = 0; s < 4; ++s) {
                    v16bf bb = load_b_frag(Wt2 + (size_t)(w * 64 + s * 16) * ND, ND, k0);
                    acc[0][s] = __builtin_amdgcn_wmma_f32_16x16x32_bf16(false, a0, false, bb,
                                                                        (short)0, acc[0][s], false, false);
                    acc[1][s] = __builtin_amdgcn_wmma_f32_16x16x32_bf16(false, a1, false, bb,
                                                                        (short)0, acc[1][s], false, false);
                }
            }
#pragma unroll
            for (int u = 0; u < 2; ++u)
#pragma unroll
                for (int s = 0; s < 4; ++s) {
                    int col = w * 64 + s * 16 + n;
                    float bias = b2[col];
#pragma unroll
                    for (int r = 0; r < 8; ++r)
                        h2s[u][(size_t)(mo + r) * LDP + col] = f2bf(lrelu(acc[u][s][r] + bias));
                }
        }
        __syncthreads();

        // ---- layer 3 + lrelu + sum over j
        {
            v8f acc[2][4] = {};
            for (int k0 = 0; k0 < ND; k0 += 32) {
                v16bf a0 = load_a_frag(h2s[0], LDP, k0);
                v16bf a1 = load_a_frag(h2s[1], LDP, k0);
#pragma unroll
                for (int s = 0; s < 4; ++s) {
                    v16bf bb = load_b_frag(Wt3 + (size_t)(w * 64 + s * 16) * ND, ND, k0);
                    acc[0][s] = __builtin_amdgcn_wmma_f32_16x16x32_bf16(false, a0, false, bb,
                                                                        (short)0, acc[0][s], false, false);
                    acc[1][s] = __builtin_amdgcn_wmma_f32_16x16x32_bf16(false, a1, false, bb,
                                                                        (short)0, acc[1][s], false, false);
                }
            }
#pragma unroll
            for (int u = 0; u < 2; ++u)
#pragma unroll
                for (int s = 0; s < 4; ++s) {
                    int col = w * 64 + s * 16 + n;
                    float bias = b3[col];
                    float psum = 0.f;
#pragma unroll
                    for (int r = 0; r < 8; ++r)
                        psum += lrelu(acc[u][s][r] + bias);      // rows mo..mo+7 at column col
                    float tot = psum + __shfl_xor(psum, 16, 32); // both lane halves -> all 16 j
                    if (lane < 16)
                        S[(size_t)(b * NV + t + u) * ND + col] = f2bf(tot);
                }
        }
        __syncthreads();
    }
}

// ---- fusion: y = Fb @ WfA + S @ WfB + bf ----------------------------------
__global__ void fusion_gemm(const unsigned short* __restrict__ Fb, const unsigned short* __restrict__ S,
                            const unsigned short* __restrict__ WtfA, const unsigned short* __restrict__ WtfB,
                            const float* __restrict__ bfv, float* __restrict__ Y) {
    int mt = blockIdx.x, nt = blockIdx.y;
    const unsigned short* Ap0 = Fb + (size_t)mt * 16 * ND;
    const unsigned short* Ap1 = S  + (size_t)mt * 16 * ND;
    v8f acc[4] = {};
    for (int k0 = 0; k0 < ND; k0 += 32) {
        v16bf a = load_a_frag(Ap0, ND, k0);
#pragma unroll
        for (int s = 0; s < 4; ++s) {
            v16bf b = load_b_frag(WtfA + (size_t)(nt * 64 + s * 16) * ND, ND, k0);
            acc[s] = __builtin_amdgcn_wmma_f32_16x16x32_bf16(false, a, false, b,
                                                             (short)0, acc[s], false, false);
        }
    }
    for (int k0 = 0; k0 < ND; k0 += 32) {
        v16bf a = load_a_frag(Ap1, ND, k0);
#pragma unroll
        for (int s = 0; s < 4; ++s) {
            v16bf b = load_b_frag(WtfB + (size_t)(nt * 64 + s * 16) * ND, ND, k0);
            acc[s] = __builtin_amdgcn_wmma_f32_16x16x32_bf16(false, a, false, b,
                                                             (short)0, acc[s], false, false);
        }
    }
    int lane = threadIdx.x & 31;
    int n    = lane & 15;
    int mo   = (lane >> 4) * 8;
#pragma unroll
    for (int s = 0; s < 4; ++s) {
        int col = nt * 64 + s * 16 + n;
        float bias = bfv[col];
#pragma unroll
        for (int r = 0; r < 8; ++r)
            Y[(size_t)(mt * 16 + mo + r) * ND + col] = acc[s][r] + bias;
    }
}

// ---- BatchNorm: deterministic 3-stage (partials -> stats -> apply) --------
// Stage 1: coalesced partial sums over (64-col x 256-row) tiles.
__global__ __launch_bounds__(256) void bn_partial(const float* __restrict__ Y,
                                                  float* __restrict__ partS,
                                                  float* __restrict__ partQ) {
    int tid = threadIdx.x;
    int tx  = tid & 63;           // column within group
    int ty  = tid >> 6;           // row phase (0..3)
    int c   = blockIdx.x * 64 + tx;
    int r0  = blockIdx.y * 256;
    float s = 0.f, q = 0.f;
    for (int r = r0 + ty; r < r0 + 256; r += 4) {
        float v = Y[(size_t)r * ND + c];
        s += v; q += v * v;
    }
    __shared__ float sred[4][64], qred[4][64];
    sred[ty][tx] = s; qred[ty][tx] = q;
    __syncthreads();
    if (ty == 0) {
        s = sred[0][tx] + sred[1][tx] + sred[2][tx] + sred[3][tx];
        q = qred[0][tx] + qred[1][tx] + qred[2][tx] + qred[3][tx];
        partS[(size_t)blockIdx.y * ND + c] = s;
        partQ[(size_t)blockIdx.y * ND + c] = q;
    }
}

// Stage 2: fixed-order combine of 16 partials -> per-column scale/shift.
__global__ void bn_stats(const float* __restrict__ partS, const float* __restrict__ partQ,
                         const float* __restrict__ gamma, const float* __restrict__ beta,
                         float* __restrict__ scale, float* __restrict__ shift) {
    int c = blockIdx.x * 256 + threadIdx.x;
    if (c >= ND) return;
    float s = 0.f, q = 0.f;
    for (int i = 0; i < 16; ++i) { s += partS[(size_t)i * ND + c]; q += partQ[(size_t)i * ND + c]; }
    float mean = s * (1.f / NROWS);
    float var  = q * (1.f / NROWS) - mean * mean;   // biased var (torch BN)
    float sc   = gamma[c] * rsqrtf(var + BN_EPS);
    scale[c] = sc;
    shift[c] = beta[c] - mean * sc;
}

// Stage 3: fully coalesced normalize + lrelu.
__global__ void bn_apply(const float* __restrict__ Y, const float* __restrict__ scale,
                         const float* __restrict__ shift, float* __restrict__ out) {
    int i = blockIdx.x * 256 + threadIdx.x;
    int c = i & (ND - 1);
    out[i] = lrelu(Y[i] * scale[c] + shift[c]);
}

// ---------------------------------------------------------------------------
extern "C" void kernel_launch(void* const* d_in, const int* in_sizes, int n_in,
                              void* d_out, int out_size, void* d_ws, size_t ws_size,
                              hipStream_t stream) {
    const float* F   = (const float*)d_in[0];
    const float* W1  = (const float*)d_in[1];
    const float* b1  = (const float*)d_in[2];
    const float* W2  = (const float*)d_in[3];
    const float* b2  = (const float*)d_in[4];
    const float* W3  = (const float*)d_in[5];
    const float* b3  = (const float*)d_in[6];
    const float* Wf  = (const float*)d_in[7];
    const float* bfv = (const float*)d_in[8];
    const float* gam = (const float*)d_in[9];
    const float* bet = (const float*)d_in[10];

    char* ws = (char*)d_ws;
    size_t off = 0;
    auto take = [&](size_t bytes) { char* p = ws + off; off += (bytes + 255) & ~size_t(255); return p; };
    unsigned short* Fb    = (unsigned short*)take((size_t)NROWS * ND * 2);  // F bf16
    unsigned short* Wt1a  = (unsigned short*)take((size_t)ND * ND * 2);
    unsigned short* Wt1b  = (unsigned short*)take((size_t)ND * ND * 2);
    unsigned short* Wt2   = (unsigned short*)take((size_t)ND * ND * 2);
    unsigned short* Wt3   = (unsigned short*)take((size_t)ND * ND * 2);
    unsigned short* WtfA  = (unsigned short*)take((size_t)ND * ND * 2);
    unsigned short* WtfB  = (unsigned short*)take((size_t)ND * ND * 2);
    float*          P     = (float*)take((size_t)NROWS * ND * 4);
    float*          Q     = (float*)take((size_t)NROWS * ND * 4);
    unsigned short* S     = (unsigned short*)take((size_t)NROWS * ND * 2);
    float*          Y     = (float*)take((size_t)NROWS * ND * 4);
    float*          partS = (float*)take((size_t)16 * ND * 4);
    float*          partQ = (float*)take((size_t)16 * ND * 4);
    float*          scale = (float*)take((size_t)ND * 4);
    float*          shift = (float*)take((size_t)ND * 4);
    (void)ws_size; // ~37 MB used

    const int ELT  = NROWS * ND;             // 2,097,152
    const int WELT = ND * ND;                // 262,144
    convert_kernel<<<(ELT + 255) / 256, 256, 0, stream>>>(F, Fb, ELT);
    transpose_kernel<<<(WELT + 255) / 256, 256, 0, stream>>>(W1, Wt1a, ND, ND, 0);
    transpose_kernel<<<(WELT + 255) / 256, 256, 0, stream>>>(W1, Wt1b, ND, ND, ND);
    transpose_kernel<<<(WELT + 255) / 256, 256, 0, stream>>>(W2, Wt2,  ND, ND, 0);
    transpose_kernel<<<(WELT + 255) / 256, 256, 0, stream>>>(W3, Wt3,  ND, ND, 0);
    transpose_kernel<<<(WELT + 255) / 256, 256, 0, stream>>>(Wf, WtfA, ND, ND, 0);
    transpose_kernel<<<(WELT + 255) / 256, 256, 0, stream>>>(Wf, WtfB, ND, ND, ND);

    // P = F @ W1_top, Q = F @ W1_bot
    gemm16x64<<<dim3(NROWS / 16, ND / 64), 32, 0, stream>>>(Fb, ND, Wt1a, ND, P, ND, ND);
    gemm16x64<<<dim3(NROWS / 16, ND / 64), 32, 0, stream>>>(Fb, ND, Wt1b, ND, Q, ND, ND);

    // fused relation MLP + neighbor sum
    relation_kernel<<<NB, 256, 0, stream>>>(P, Q, Wt2, Wt3, b1, b2, b3, S);

    // fusion linear
    fusion_gemm<<<dim3(NROWS / 16, ND / 64), 32, 0, stream>>>(Fb, S, WtfA, WtfB, bfv, Y);

    // batchnorm + lrelu -> output
    bn_partial<<<dim3(ND / 64, NROWS / 256), 256, 0, stream>>>(Y, partS, partQ);
    bn_stats<<<(ND + 255) / 256, 256, 0, stream>>>(partS, partQ, gam, bet, scale, shift);
    bn_apply<<<ELT / 256, 256, 0, stream>>>(Y, scale, shift, (float*)d_out);
}
